// MIKT_19542101196824
// MI455X (gfx1250) — matrix operand order repaired and checked by reading, not compiled, not run on hardware
//
#include <hip/hip_runtime.h>
#include <cmath>

#define DI __device__ __forceinline__

typedef __bf16 bf16_t;
typedef __attribute__((ext_vector_type(16))) __bf16 v16bf;
typedef __attribute__((ext_vector_type(8)))  float  v8f;
typedef __attribute__((ext_vector_type(4))) unsigned int u32x4;
typedef __attribute__((ext_vector_type(8))) int i32x8;
typedef __attribute__((ext_vector_type(4))) int i32x4;

constexpr int BATCH = 128;
constexpr int L     = 201;
constexpr int T     = 200;   // L-1 scan steps
constexpr int S     = 300;
constexpr int D     = 128;

DI float sigm(float x) { return 1.0f / (1.0f + expf(-x)); }

DI void softmax_inplace(float* v, int n) {
  float mx = -1e30f;
  for (int i = 0; i < n; ++i) mx = fmaxf(mx, v[i]);
  float s = 0.0f;
  for (int i = 0; i < n; ++i) { float e = expf(v[i] - mx); v[i] = e; s += e; }
  float inv = 1.0f / s;
  for (int i = 0; i < n; ++i) v[i] *= inv;
}

// A-fragment gather for v_wmma_*_16x16x32_bf16, per ISA 16-bit A 16x32 layout:
// lanes 0-15: M=lane, VGPR v<4 -> K=2v,2v+1 ; v>=4 -> K=16+2(v-4)..
// lanes 16-31: same M, K offset +8. Lowers to two ds_load_b128 per fragment.
DI v16bf load_a_frag(const bf16_t* base, int rowStride, int kt, int lane) {
  const int m    = lane & 15;
  const int hadd = (lane >= 16) ? 8 : 0;
  v16bf a;
#pragma unroll
  for (int v = 0; v < 8; ++v) {
    int kb = (v < 4) ? (2 * v + hadd) : (16 + 2 * (v - 4) + hadd);
    int k  = kt * 32 + kb;
    a[2 * v]     = base[m * rowStride + k];
    a[2 * v + 1] = base[m * rowStride + k + 1];
  }
  return a;
}

// ---------- precompute kernels ----------

// qs = skill_lin(skill_embed) : one block per skill row
__global__ void k_qs(const float* __restrict__ skill_embed,
                     const float* __restrict__ W, const float* __restrict__ bvec,
                     float* __restrict__ qs) {
  __shared__ float se[D];
  const int s = blockIdx.x, d = threadIdx.x;
  se[d] = skill_embed[s * D + d];
  __syncthreads();
  float acc = bvec[d];
  for (int k = 0; k < D; ++k) acc += W[d * D + k] * se[k];
  qs[s * D + d] = acc;
}

// c_af = af1_W[:, D:2D] @ time_state[1] + af1_b   (constant across the scan)
__global__ void k_caf(const float* __restrict__ af1_W, const float* __restrict__ af1_b,
                      const float* __restrict__ time_state, float* __restrict__ caf) {
  const int d = threadIdx.x;
  float acc = af1_b[d];
  for (int k = 0; k < D; ++k) acc += af1_W[d * 2 * D + D + k] * time_state[D + k];
  caf[d] = acc;
}

// Per-(b,t) state-independent precompute: npe (pe), tg, obt, diff.
__global__ void __launch_bounds__(128)
k_pre(const int* __restrict__ questions, const int* __restrict__ responses,
      const float* __restrict__ p2s,
      const float* __restrict__ pro_embed, const float* __restrict__ skill_embed,
      const float* __restrict__ pro_diff_w,
      const float* __restrict__ pro_lin_W, const float* __restrict__ pro_lin_b,
      const float* __restrict__ pro_change_W, const float* __restrict__ pro_change_b,
      const float* __restrict__ no1_W, const float* __restrict__ no1_b,
      const float* __restrict__ no2_W, const float* __restrict__ no2_b,
      const float* __restrict__ ao_W, const float* __restrict__ ao_b,
      const float* __restrict__ ans_embed, const float* __restrict__ qsbuf,
      float* __restrict__ ws_pe, float* __restrict__ ws_tg,
      float* __restrict__ ws_obt, float* __restrict__ ws_diff) {
  __shared__ float qp[D], sm[D], xin[D], t1[D], aw[304];
  __shared__ int act[304], cnts;
  const int blk = blockIdx.x;
  const int b = blk / T, t = blk % T;
  const int d = threadIdx.x;
  const int pro = questions[b * L + t + 1];
  int resp = responses[b * L + t + 1];
  const int na = (resp > -1) ? resp : 0;   // masked_r

  {
    float acc = pro_lin_b[d];
    const float* w  = pro_lin_W + d * D;
    const float* pe = pro_embed + (size_t)pro * D;
    for (int k = 0; k < D; ++k) acc += w[k] * pe[k];
    qp[d] = acc;
  }
  if (d == 0) {   // deterministic active-skill compaction
    const float* row = p2s + (size_t)pro * S;
    int c = 0;
    for (int s = 0; s < S; ++s)
      if (row[s] != 0.0f) act[c++] = s;
    cnts = c;
  }
  __syncthreads();
  const int c = cnts;
  for (int i = d; i < c; i += D) {
    const float* qr = qsbuf + act[i] * D;
    float acc = 0.0f;
    for (int k = 0; k < D; ++k) acc += qp[k] * qr[k];
    aw[i] = acc * 0.088388347648318447f;   // 1/sqrt(128)
  }
  __syncthreads();
  if (d == 0) softmax_inplace(aw, c);
  __syncthreads();
  float pes = 0.0f, sms = 0.0f;
  for (int i = 0; i < c; ++i) {
    float e = skill_embed[act[i] * D + d];
    pes += aw[i] * e;
    sms += e;
  }
  sm[d] = sms / ((float)c + 1e-8f);
  __syncthreads();
  float pc = pro_change_b[d];
  {
    const float* w = pro_change_W + d * D;
    for (int k = 0; k < D; ++k) pc += w[k] * sm[k];
  }
  const float diff = sigm(pro_diff_w[pro]);
  const float npe = pes + diff * pc;
  ws_pe[(size_t)blk * D + d] = npe;
  xin[d] = npe + ans_embed[na * D + d];
  __syncthreads();
  {
    float acc = no1_b[d];
    const float* w = no1_W + d * D;
    for (int k = 0; k < D; ++k) acc += w[k] * xin[k];
    t1[d] = tanhf(acc);
  }
  __syncthreads();
  {
    float acc = no2_b[d];
    const float* w = no2_W + d * D;
    for (int k = 0; k < D; ++k) acc += w[k] * t1[k];
    ws_tg[(size_t)blk * D + d] = tanhf(tanhf(acc));
  }
  {
    float acc = ao_b[d];
    const float* w = ao_W + d * D;
    for (int k = 0; k < D; ++k) acc += w[k] * xin[k];
    ws_obt[(size_t)blk * D + d] = tanhf(acc);
  }
  if (d == 0) ws_diff[blk] = diff;
}

// ---------- persistent scan kernel: one workgroup per batch element ----------
// LDS-resident carry (ss 300x128 f32 = 150KB fits the 320KB WGP LDS),
// sparse active-skill gate MLP via v_wmma_f32_16x16x32_bf16,
// carry state staged into LDS via the Tensor Data Mover.

__global__ void __launch_bounds__(256, 1)
k_scan(const int* __restrict__ questions, const float* __restrict__ p2s,
       const float* __restrict__ time_state,
       const float* __restrict__ skill_state0, const float* __restrict__ all_state0,
       const float* __restrict__ af1_W, const float* __restrict__ af2_W,
       const float* __restrict__ af2_b,
       const float* __restrict__ sf1_W, const float* __restrict__ sf1_b,
       const float* __restrict__ sf2_W, const float* __restrict__ sf2_b,
       const float* __restrict__ pattn_W, const float* __restrict__ pattn_b,
       const float* __restrict__ pa1_W, const float* __restrict__ pa1_b,
       const float* __restrict__ pa2_W, const float* __restrict__ pa2_b,
       const float* __restrict__ ws_pe, const float* __restrict__ ws_tg,
       const float* __restrict__ ws_obt, const float* __restrict__ ws_diff,
       const float* __restrict__ ws_caf,
       float* __restrict__ out) {
  extern __shared__ char smem[];
  float* ssl  = (float*)smem;       // S*D carry state (persistent across scan)
  float* astv = ssl + S * D;        // D
  float* fav  = astv + D;
  float* r1v  = fav + D;
  float* hfav = r1v + D;
  float* pev  = hfav + D;
  float* tgl  = pev + D;
  float* nnsv = tgl + D;
  float* aav  = nnsv + D;
  float* cat2 = aav + D;            // 3D
  float* h2v  = cat2 + 3 * D;       // D
  float* lg1  = h2v + D;            // 304
  float* lg2  = lg1 + 304;          // 304
  int*   act  = (int*)(lg2 + 304);  // 304
  int*   lstv = act + 304;          // S
  int*   meta = lstv + S;           // [0]=cnt [1]=pro
  bf16_t* Abuf = (bf16_t*)(meta + 4); // 16 x 256
  bf16_t* Hbuf = Abuf + 16 * 256;     // 16 x 128

  const int b    = blockIdx.x;
  const int tid  = threadIdx.x;
  const int wave = tid >> 5;
  const int lane = tid & 31;
  const int n16  = lane & 15;
  const bool hi  = lane >= 16;
  const int oc   = wave * 16 + n16;   // output column owned by this wave/lane

  // Preload WMMA B fragments (weights) in registers, once per kernel.
  // 16-bit B 32x16 layout: lanes 0-15 N=lane, VGPR v<4 -> K=2v,2v+1 ; v>=4 -> K=16+2(v-4)..; lanes 16-31 K+=8.
  v16bf Bs1[8], Bs2[4];
#pragma unroll
  for (int kt = 0; kt < 8; ++kt) {
#pragma unroll
    for (int v = 0; v < 8; ++v) {
      int k = kt * 32 + (hi ? 8 : 0) + ((v < 4) ? 2 * v : 16 + 2 * (v - 4));
      Bs1[kt][2 * v]     = (bf16_t)sf1_W[oc * (3 * D) + k];
      Bs1[kt][2 * v + 1] = (bf16_t)sf1_W[oc * (3 * D) + k + 1];
    }
  }
#pragma unroll
  for (int kt = 0; kt < 4; ++kt) {
#pragma unroll
    for (int v = 0; v < 8; ++v) {
      int k = kt * 32 + (hi ? 8 : 0) + ((v < 4) ? 2 * v : 16 + 2 * (v - 4));
      Bs2[kt][2 * v]     = (bf16_t)sf2_W[oc * D + k];
      Bs2[kt][2 * v + 1] = (bf16_t)sf2_W[oc * D + k + 1];
    }
  }
  const float cinit2 = sf2_b[oc];

  // ---- init carry: TDM bulk HBM->LDS of ss, then (idempotent) fill ----
#if __has_builtin(__builtin_amdgcn_tensor_load_to_lds)
  if (tid < 32) {   // wave 0 issues one Tensor Data Mover descriptor
    const uint64_t ga      = (uint64_t)(uintptr_t)skill_state0;
    const uint32_t lds_off = (uint32_t)(uintptr_t)ssl;     // LDS byte address
    const uint32_t nel     = (uint32_t)(S * D);            // 38400 f32 elements
    u32x4 g0;
    g0[0] = 1u;                                            // count=1, user desc
    g0[1] = lds_off;                                       // lds_addr
    g0[2] = (uint32_t)(ga & 0xFFFFFFFFu);                  // global_addr lo
    g0[3] = (uint32_t)((ga >> 32) & 0x01FFFFFFu) | (2u << 30); // addr hi | type=2
    i32x8 g1;
    g1[0] = (int)(2u << 16);                  // wg_mask=0, data_size=2 (4B)
    g1[1] = (int)((nel & 0xFFFFu) << 16);     // tensor_dim0[15:0]
    g1[2] = (int)((nel >> 16) | (1u << 16));  // tensor_dim0[31:16], tensor_dim1=1
    g1[3] = (int)((nel & 0xFFFFu) << 16);     // tile_dim0 = nel (1-D tile)
    g1[4] = 0;                                // tile_dim1=0, tile_dim2=0
    g1[5] = (int)nel;                         // tensor_dim0_stride lo32
    g1[6] = 0;
    g1[7] = 0;
    i32x4 g2 = {0, 0, 0, 0};
    i32x4 g3 = {0, 0, 0, 0};
    i32x8 g4 = {0, 0, 0, 0, 0, 0, 0, 0};
    __builtin_amdgcn_tensor_load_to_lds(g0, g1, g2, g3, g4, 0);
    __builtin_amdgcn_s_wait_tensorcnt(0);
  }
  __syncthreads();
#endif
  for (int idx = tid; idx < S * D; idx += 256) ssl[idx] = skill_state0[idx];
  if (tid < D) astv[tid] = all_state0[tid];
  for (int s = tid; s < S; s += 256) lstv[s] = 0;
  __syncthreads();

  for (int t = 0; t < T; ++t) {
    const int base = (b * T + t) * D;
    if (tid < D) { pev[tid] = ws_pe[base + tid]; tgl[tid] = ws_tg[base + tid]; }
    if (tid == 0) { meta[1] = questions[b * L + t + 1]; meta[0] = 0; }
    __syncthreads();
    const int pro = meta[1];

    // fa stage 1: r1 = relu(af1_W[:, :D] @ ast + c_af)
    if (tid < D) {
      float acc = ws_caf[tid];
      const float* w = af1_W + tid * (2 * D);
      for (int k = 0; k < D; ++k) acc += w[k] * astv[k];
      r1v[tid] = fmaxf(acc, 0.0f);
    }
    // deterministic active-skill compaction on an otherwise idle thread
    if (tid == 255) {
      const float* row = p2s + (size_t)pro * S;
      int c = 0;
      for (int s = 0; s < S; ++s)
        if (row[s] != 0.0f) act[c++] = s;
      meta[0] = c;
    }
    __syncthreads();

    // fa stage 2
    if (tid < D) {
      float acc = af2_b[tid];
      const float* w = af2_W + tid * D;
      for (int k = 0; k < D; ++k) acc += w[k] * r1v[k];
      fav[tid] = astv[tid] * sigm(acc);
    }
    __syncthreads();

    // hfa = sf1_W[:, 2D:3D] @ fa + sf1_b  (WMMA stage-1 accumulator init)
    if (tid < D) {
      float acc = sf1_b[tid];
      const float* w = sf1_W + tid * (3 * D) + 2 * D;
      for (int k = 0; k < D; ++k) acc += w[k] * fav[k];
      hfav[tid] = acc;
    }
    __syncthreads();

    const int cnt    = meta[0];
    const int ntiles = (cnt + 15) >> 4;

    for (int tile = 0; tile < ntiles; ++tile) {
      // stage A rows [ss | stge] (bf16), zero-padded past cnt
      for (int idx = tid; idx < 16 * 256; idx += 256) {
        int r = idx >> 8, k = idx & 255;
        int i = tile * 16 + r;
        float val = 0.0f;
        if (i < cnt) {
          int s = act[i];
          val = (k < D) ? ssl[s * D + k]
                        : time_state[(size_t)(t - lstv[s]) * D + (k - D)];
        }
        Abuf[idx] = (bf16_t)val;
      }
      __syncthreads();

      // stage 1: h = relu(A @ sf1_W[:, :2D]^T + hfa)   (K = 256)
      v8f c;
#pragma unroll
      for (int r = 0; r < 8; ++r) c[r] = hfav[oc];
#pragma unroll
      for (int kt = 0; kt < 8; ++kt) {
        v16bf a = load_a_frag(Abuf, 256, kt, lane);
        c = __builtin_amdgcn_wmma_f32_16x16x32_bf16(false, a, false, Bs1[kt],
                                                    (short)0, c, false, false);
      }
#pragma unroll
      for (int r = 0; r < 8; ++r) {
        int m = r + (hi ? 8 : 0);
        Hbuf[m * D + oc] = (bf16_t)fmaxf(c[r], 0.0f);
      }
      __syncthreads();

      // stage 2: sf = sigmoid(h @ sf2_W^T + b); ss *= sf   (K = 128)
#pragma unroll
      for (int r = 0; r < 8; ++r) c[r] = cinit2;
#pragma unroll
      for (int kt = 0; kt < 4; ++kt) {
        v16bf a = load_a_frag(Hbuf, D, kt, lane);
        c = __builtin_amdgcn_wmma_f32_16x16x32_bf16(false, a, false, Bs2[kt],
                                                    (short)0, c, false, false);
      }
#pragma unroll
      for (int r = 0; r < 8; ++r) {
        int m = r + (hi ? 8 : 0);
        int i = tile * 16 + m;
        if (i < cnt) ssl[act[i] * D + oc] *= sigm(c[r]);
      }
      __syncthreads();
    }

    // attention logits over active skills (post-decay ss)
    for (int i = tid; i < cnt; i += 256) {
      const float* sr = ssl + act[i] * D;
      float a1 = 0.0f, a2 = 0.0f;
      for (int d = 0; d < D; ++d) { float v = sr[d]; a1 += pev[d] * v; a2 += tgl[d] * v; }
      lg1[i] = a1 * (1.0f / 128.0f);
      lg2[i] = a2 * (1.0f / 128.0f);
    }
    __syncthreads();
    if (tid == 0)  softmax_inplace(lg1, cnt);
    if (tid == 32) softmax_inplace(lg2, cnt);
    __syncthreads();

    if (tid < D) {
      float acc = 0.0f;
      for (int i = 0; i < cnt; ++i) acc += lg1[i] * ssl[act[i] * D + tid];
      nnsv[tid] = acc;
    }
    __syncthreads();

    // aa = sigmoid(pattn([nns, fa, pe]))
    if (tid < D) {
      float acc = pattn_b[tid];
      const float* w = pattn_W + tid * (3 * D);
      for (int k = 0; k < D; ++k) acc += w[k] * nnsv[k];
      for (int k = 0; k < D; ++k) acc += w[D + k] * fav[k];
      for (int k = 0; k < D; ++k) acc += w[2 * D + k] * pev[k];
      aav[tid] = sigm(acc);
    }
    __syncthreads();
    if (tid < D) {
      cat2[tid]         = (1.0f - aav[tid]) * nnsv[tid];
      cat2[D + tid]     = aav[tid] * fav[tid];
      cat2[2 * D + tid] = pev[tid];
    }
    __syncthreads();
    if (tid < D) {
      float acc = pa1_b[tid];
      const float* w = pa1_W + tid * (3 * D);
      for (int k = 0; k < 3 * D; ++k) acc += w[k] * cat2[k];
      h2v[tid] = fmaxf(acc, 0.0f);
    }
    __syncthreads();
    if (tid == 0) {
      float acc = pa2_b[0];
      for (int k = 0; k < D; ++k) acc += pa2_W[k] * h2v[k];
      float ability = sigm(acc);
      out[b * T + t] = sigm(5.0f * (ability - ws_diff[b * T + t]));
    }

    // carry updates for next step
    for (int idx = tid; idx < cnt * D; idx += 256) {
      int i = idx >> 7, d = idx & (D - 1);
      ssl[act[i] * D + d] += lg2[i] * tgl[d];
    }
    for (int i = tid; i < cnt; i += 256) lstv[act[i]] = t;
    if (tid < D) astv[tid] = fav[tid] + ws_obt[base + tid];

    if (t + 1 < T) {   // pull next-step operand rows toward the WGP
      if (tid == 0)  __builtin_prefetch(ws_pe  + base + D, 0, 0);
      if (tid == 32) __builtin_prefetch(ws_tg  + base + D, 0, 0);
      if (tid == 64) __builtin_prefetch(ws_obt + base + D, 0, 0);
    }
    __syncthreads();
  }
}

extern "C" void kernel_launch(void* const* d_in, const int* in_sizes, int n_in,
                              void* d_out, int out_size, void* d_ws, size_t ws_size,
                              hipStream_t stream) {
  const int*   questions    = (const int*)  d_in[0];
  const int*   responses    = (const int*)  d_in[1];
  const float* pro2skill    = (const float*)d_in[2];
  const float* pro_embed    = (const float*)d_in[3];
  const float* skill_embed  = (const float*)d_in[4];
  const float* pro_diff_w   = (const float*)d_in[5];
  const float* pro_lin_W    = (const float*)d_in[6];
  const float* pro_lin_b    = (const float*)d_in[7];
  const float* skill_lin_W  = (const float*)d_in[8];
  const float* skill_lin_b  = (const float*)d_in[9];
  const float* pro_change_W = (const float*)d_in[10];
  const float* pro_change_b = (const float*)d_in[11];
  const float* all_obtain_W = (const float*)d_in[12];
  const float* all_obtain_b = (const float*)d_in[13];
  const float* af1_W = (const float*)d_in[14];
  const float* af1_b = (const float*)d_in[15];
  const float* af2_W = (const float*)d_in[16];
  const float* af2_b = (const float*)d_in[17];
  const float* no1_W = (const float*)d_in[18];
  const float* no1_b = (const float*)d_in[19];
  const float* no2_W = (const float*)d_in[20];
  const float* no2_b = (const float*)d_in[21];
  const float* pa1_W = (const float*)d_in[22];
  const float* pa1_b = (const float*)d_in[23];
  const float* pa2_W = (const float*)d_in[24];
  const float* pa2_b = (const float*)d_in[25];
  const float* sf1_W = (const float*)d_in[26];
  const float* sf1_b = (const float*)d_in[27];
  const float* sf2_W = (const float*)d_in[28];
  const float* sf2_b = (const float*)d_in[29];
  const float* pattn_W = (const float*)d_in[30];
  const float* pattn_b = (const float*)d_in[31];
  const float* ans_embed    = (const float*)d_in[32];
  const float* time_state   = (const float*)d_in[33];
  const float* all_state0   = (const float*)d_in[34];
  const float* skill_state0 = (const float*)d_in[35];
  (void)in_sizes; (void)n_in; (void)out_size; (void)ws_size;

  float* ws      = (float*)d_ws;
  float* ws_pe   = ws;
  float* ws_tg   = ws_pe   + (size_t)BATCH * T * D;
  float* ws_obt  = ws_tg   + (size_t)BATCH * T * D;
  float* ws_diff = ws_obt  + (size_t)BATCH * T * D;
  float* ws_qs   = ws_diff + (size_t)BATCH * T;
  float* ws_caf  = ws_qs   + (size_t)S * D;

  k_qs<<<S, D, 0, stream>>>(skill_embed, skill_lin_W, skill_lin_b, ws_qs);
  k_caf<<<1, D, 0, stream>>>(af1_W, af1_b, time_state, ws_caf);
  k_pre<<<BATCH * T, D, 0, stream>>>(questions, responses, pro2skill,
        pro_embed, skill_embed, pro_diff_w,
        pro_lin_W, pro_lin_b, pro_change_W, pro_change_b,
        no1_W, no1_b, no2_W, no2_b, all_obtain_W, all_obtain_b,
        ans_embed, ws_qs, ws_pe, ws_tg, ws_obt, ws_diff);

  const size_t lds_bytes =
      (size_t)(S * D + 8 * D + 3 * D + D + 2 * 304) * 4   // floats
    + (size_t)(304 + S + 4) * 4                           // ints
    + (size_t)(16 * 256 + 16 * 128) * 2;                  // bf16 staging
  (void)hipFuncSetAttribute((const void*)k_scan,
                            hipFuncAttributeMaxDynamicSharedMemorySize,
                            (int)lds_bytes);
  k_scan<<<BATCH, 256, lds_bytes, stream>>>(questions, pro2skill, time_state,
        skill_state0, all_state0,
        af1_W, af2_W, af2_b,
        sf1_W, sf1_b, sf2_W, sf2_b,
        pattn_W, pattn_b, pa1_W, pa1_b, pa2_W, pa2_b,
        ws_pe, ws_tg, ws_obt, ws_diff, ws_caf, (float*)d_out);
}
